// TrajPreLocalAttnLong_1786706395397
// MI455X (gfx1250) — matrix operand order repaired and checked by reading, not compiled, not run on hardware
//
#include <hip/hip_runtime.h>
#include <math.h>

typedef __attribute__((ext_vector_type(16))) _Float16 v16h;
typedef __attribute__((ext_vector_type(8)))  _Float16 v8h;
typedef __attribute__((ext_vector_type(8)))  float    v8f;

union Frag16 { v16h v; v8h h[2]; };

// ---- problem constants ----
#define N_TOT   8192
#define TLEN    2048
#define NHIST   6144
#define LOC_SZ  50000
#define LOC_E   512
#define TIM_E   32
#define HID     512
#define INP     544     // 512 + 32, = 17*32
#define G3      1536    // 3*HID
#define OUT2    1024    // 2*HID

// =====================================================================
// 1. Embedding gather + concat + f32->f16
// =====================================================================
__global__ void embed_kernel(const int* __restrict__ loc, const int* __restrict__ tim,
                             const float* __restrict__ embL, const float* __restrict__ embT,
                             _Float16* __restrict__ X)
{
    int idx = blockIdx.x * blockDim.x + threadIdx.x;
    if (idx >= N_TOT * INP) return;
    int i = idx / INP, j = idx - i * INP;
    float v;
    if (j < LOC_E) v = embL[(size_t)loc[i] * LOC_E + j];
    else           v = embT[(size_t)tim[i] * TIM_E + (j - LOC_E)];
    X[idx] = (_Float16)v;
}

// =====================================================================
// 2. Transposes (operand prep: out[c][r] = in[r][c])
// =====================================================================
__global__ void transpose_f32_to_f16(const float* __restrict__ in, _Float16* __restrict__ out,
                                     int R, int C)
{
    int idx = blockIdx.x * blockDim.x + threadIdx.x;
    if (idx >= R * C) return;
    int r = idx / C, c = idx - r * C;
    out[(size_t)c * R + r] = (_Float16)in[idx];
}

__global__ void transpose_f16(const _Float16* __restrict__ in, _Float16* __restrict__ out,
                              int R, int C)
{
    int idx = blockIdx.x * blockDim.x + threadIdx.x;
    if (idx >= R * C) return;
    int r = idx / C, c = idx - r * C;
    out[(size_t)c * R + r] = in[idx];
}

// =====================================================================
// 3. WMMA NT-GEMM: C[M][N](f32) = A[M][K](f16,row-major) * Bt[N][K](f16,row-major)^T (+bias[N])
//    Requirements: M%16==0, N%16==0, K%32==0, rows 16B-aligned (lda/ldb%8==0).
//    Block = 128 threads (4 waves); each wave computes a 16x64 strip (4 tiles).
//    Wave id / tile base are forced scalar (readfirstlane) so ALL control flow
//    is s_cbranch (EXEC stays all-ones around every v_wmma, per ISA rule).
//    Ragged N edge handled by clamping the tile index (redundant compute),
//    with stores skipped by a scalar guard.
// =====================================================================
__global__ __launch_bounds__(128)
void gemm_nt_wmma(const _Float16* __restrict__ A, int lda,
                  const _Float16* __restrict__ Bt, int ldb,
                  float* __restrict__ C, int ldc,
                  const float* __restrict__ bias,
                  int M, int N, int K)
{
    const int lane = threadIdx.x & 31;
    const int wave = __builtin_amdgcn_readfirstlane(threadIdx.x >> 5);  // scalar
    const int r    = lane & 15;    // M-row (A frag) / N-col (B frag, C store)
    const int g    = lane >> 4;    // lane-group (K half selector)
    const int nTiles = N >> 4;
    const int ntBase = (blockIdx.x * 4 + wave) * 4;                     // scalar
    if (ntBase >= nTiles) return;                                       // scalar branch
    const int mBase = blockIdx.y << 4;

    // Clamp edge tiles to the last valid tile (recomputed, stores guarded).
    int ntc[4];
#pragma unroll
    for (int t = 0; t < 4; ++t) {
        int nt = ntBase + t;
        ntc[t] = (nt < nTiles) ? nt : (nTiles - 1);
    }

    const _Float16* aRow = A + (size_t)(mBase + r) * lda;
    const _Float16* bRow[4];
#pragma unroll
    for (int t = 0; t < 4; ++t)
        bRow[t] = Bt + (size_t)(ntc[t] * 16 + r) * ldb;

    v8f acc[4] = {};

    for (int k0 = 0; k0 < K; k0 += 32) {
        // A fragment (16x32 f16): lane holds row m, K = k0+g*8..+7 and k0+16+g*8..+7
        Frag16 a;
        a.h[0] = *(const v8h*)(aRow + k0 + g * 8);
        a.h[1] = *(const v8h*)(aRow + k0 + 16 + g * 8);
        __builtin_prefetch(aRow + k0 + 32, 0, 0);      // global_prefetch_b8

        // B fragments (32x16 f16): lane holds col n = nt*16+r, K = k0+g*8..+7 and k0+16+g*8..+7
        Frag16 b[4];
#pragma unroll
        for (int t = 0; t < 4; ++t) {
            const _Float16* bp = bRow[t] + k0 + g * 8;
            b[t].h[0] = *(const v8h*)(bp);
            b[t].h[1] = *(const v8h*)(bp + 16);
        }

        // 4 back-to-back WMMAs sharing the A fragment, EXEC untouched.
#pragma unroll
        for (int t = 0; t < 4; ++t)
            acc[t] = __builtin_amdgcn_wmma_f32_16x16x32_f16(
                         false, a.v, false, b[t].v, (short)0, acc[t], false, false);
    }

#pragma unroll
    for (int t = 0; t < 4; ++t) {
        if (ntBase + t >= nTiles) break;               // scalar branch
        const int col = ntc[t] * 16 + r;
        const float bv = bias ? bias[col] : 0.0f;
        // C/D layout: VGPR j holds M = mBase + g*8 + j, N = col
        float* cp = C + (size_t)(mBase + g * 8) * ldc + col;
#pragma unroll
        for (int j = 0; j < 8; ++j) cp[(size_t)j * ldc] = acc[t][j] + bv;
    }
}

// =====================================================================
// 4. GRU zero-state gate nonlinearity:
//    h = (1-sigmoid(gz+bz)) * tanh(gh + sigmoid(gr+br)*bh)
// =====================================================================
__global__ void gru_gate_kernel(const float* __restrict__ gates,  // [rows][1536] (x@W + b_in)
                                const float* __restrict__ brec,   // [1536]
                                _Float16* __restrict__ H,         // [rows][512]
                                _Float16* __restrict__ outLeft,   // optional: [rows][1024] left half
                                int rows)
{
    int idx = blockIdx.x * blockDim.x + threadIdx.x;
    if (idx >= rows * HID) return;
    int i = idx >> 9, n = idx & (HID - 1);
    const float* gp = gates + (size_t)i * G3;
    float z  = 1.0f / (1.0f + __expf(-(gp[n]           + brec[n])));
    float rg = 1.0f / (1.0f + __expf(-(gp[HID + n]     + brec[HID + n])));
    float hh = tanhf(gp[2 * HID + n] + rg * brec[2 * HID + n]);
    float h  = (1.0f - z) * hh;
    H[idx] = (_Float16)h;
    if (outLeft) outLeft[(size_t)i * OUT2 + n] = (_Float16)h;
}

// =====================================================================
// 5. Row softmax: f32 in -> f16 out (attention), one block per row
// =====================================================================
__global__ __launch_bounds__(256)
void softmax_f32_to_f16(const float* __restrict__ in, _Float16* __restrict__ out, int cols)
{
    const int tid = threadIdx.x;
    const float* x = in + (size_t)blockIdx.x * cols;
    _Float16* y    = out + (size_t)blockIdx.x * cols;
    __shared__ float sm[256], ss[256];

    float m = -INFINITY, s = 0.0f;
    for (int c = tid; c < cols; c += blockDim.x) {
        float v = x[c];
        float mN = fmaxf(m, v);
        s = s * __expf(m - mN) + __expf(v - mN);
        m = mN;
    }
    sm[tid] = m; ss[tid] = s;
    __syncthreads();
    for (int st = 128; st > 0; st >>= 1) {
        if (tid < st) {
            float m2 = sm[tid + st], s2 = ss[tid + st];
            float mN = fmaxf(sm[tid], m2);
            ss[tid] = ss[tid] * __expf(sm[tid] - mN) + s2 * __expf(m2 - mN);
            sm[tid] = mN;
        }
        __syncthreads();
    }
    m = sm[0];
    float inv = 1.0f / ss[0];
    for (int c = tid; c < cols; c += blockDim.x)
        y[c] = (_Float16)(__expf(x[c] - m) * inv);
}

// =====================================================================
// 6. Row softmax in place (final scores), online max/sum to save one pass
// =====================================================================
__global__ __launch_bounds__(256)
void softmax_f32_inplace(float* __restrict__ data, int cols)
{
    const int tid = threadIdx.x;
    float* x = data + (size_t)blockIdx.x * cols;
    __shared__ float sm[256], ss[256];

    float m = -INFINITY, s = 0.0f;
    for (int c = tid; c < cols; c += blockDim.x) {
        float v = x[c];
        float mN = fmaxf(m, v);
        s = s * __expf(m - mN) + __expf(v - mN);
        m = mN;
    }
    sm[tid] = m; ss[tid] = s;
    __syncthreads();
    for (int st = 128; st > 0; st >>= 1) {
        if (tid < st) {
            float m2 = sm[tid + st], s2 = ss[tid + st];
            float mN = fmaxf(sm[tid], m2);
            ss[tid] = ss[tid] * __expf(sm[tid] - mN) + s2 * __expf(m2 - mN);
            sm[tid] = mN;
        }
        __syncthreads();
    }
    m = sm[0];
    float inv = 1.0f / ss[0];
    for (int c = tid; c < cols; c += blockDim.x)
        x[c] = __expf(x[c] - m) * inv;
}

// =====================================================================
// 7. Pack context (f32) into right half of OUT (f16)
// =====================================================================
__global__ void pack_out_right(const float* __restrict__ ctx, _Float16* __restrict__ OUT)
{
    int idx = blockIdx.x * blockDim.x + threadIdx.x;
    if (idx >= TLEN * HID) return;
    int i = idx >> 9, n = idx & (HID - 1);
    OUT[(size_t)i * OUT2 + HID + n] = (_Float16)ctx[idx];
}

// =====================================================================
// launcher
// =====================================================================
extern "C" void kernel_launch(void* const* d_in, const int* in_sizes, int n_in,
                              void* d_out, int out_size, void* d_ws, size_t ws_size,
                              hipStream_t stream)
{
    const int*   loc     = (const int*)d_in[0];
    const int*   tim     = (const int*)d_in[1];
    const float* emb_loc = (const float*)d_in[2];
    const float* emb_tim = (const float*)d_in[3];
    const float* W_enc   = (const float*)d_in[4];
    const float* bi_enc  = (const float*)d_in[5];
    const float* bh_enc  = (const float*)d_in[6];
    const float* W_dec   = (const float*)d_in[7];
    const float* bi_dec  = (const float*)d_in[8];
    const float* bh_dec  = (const float*)d_in[9];
    const float* W_fc    = (const float*)d_in[10];
    const float* b_fc    = (const float*)d_in[11];
    float*       score   = (float*)d_out;

    // -------- workspace layout (bytes) --------
    char* ws = (char*)d_ws;
    size_t off = 0;
    auto alloc = [&](size_t bytes) { char* p = ws + off; off += (bytes + 255) & ~(size_t)255; return p; };
    _Float16* X     = (_Float16*)alloc((size_t)N_TOT * INP * 2);       //  8.9 MB
    _Float16* WencT = (_Float16*)alloc((size_t)G3 * INP * 2);          //  1.7 MB  [1536][544]
    _Float16* WdecT = (_Float16*)alloc((size_t)G3 * INP * 2);          //  1.7 MB
    _Float16* H     = (_Float16*)alloc((size_t)N_TOT * HID * 2);       //  8.4 MB  hist rows 0..6143, cur 6144..8191
    float*    SCR   = (float*)   alloc((size_t)N_TOT * G3 * 4);        // 50.3 MB  gates -> attn logits -> context
    _Float16* ATTN  = (_Float16*)alloc((size_t)TLEN * NHIST * 2);      // 25.2 MB
    _Float16* HHT   = (_Float16*)alloc((size_t)HID * NHIST * 2);       //  6.3 MB  [512][6144]
    _Float16* OUT   = (_Float16*)alloc((size_t)TLEN * OUT2 * 2);       //  4.2 MB
    _Float16* WfcT  = (_Float16*)alloc((size_t)LOC_SZ * OUT2 * 2);     //102.4 MB  [50000][1024]
    (void)ws_size; (void)in_sizes; (void)n_in; (void)out_size;

    const int T = 256;

    // 1. embedding gather -> X f16 [8192][544]
    embed_kernel<<<(N_TOT * INP + T - 1) / T, T, 0, stream>>>(loc, tim, emb_loc, emb_tim, X);

    // 2. operand prep (transpose + f16 convert)
    transpose_f32_to_f16<<<(INP * G3 + T - 1) / T, T, 0, stream>>>(W_enc, WencT, INP, G3);
    transpose_f32_to_f16<<<(INP * G3 + T - 1) / T, T, 0, stream>>>(W_dec, WdecT, INP, G3);
    transpose_f32_to_f16<<<(OUT2 * LOC_SZ + T - 1) / T, T, 0, stream>>>(W_fc, WfcT, OUT2, LOC_SZ);

    // 3. GRU gate GEMMs: gates = X @ W + b_in   (K=544, N=1536)
    {
        dim3 blk(128);
        dim3 grdE((G3 / 16 + 15) / 16, NHIST / 16);
        gemm_nt_wmma<<<grdE, blk, 0, stream>>>(X, INP, WencT, INP, SCR, G3, bi_enc,
                                               NHIST, G3, INP);
        dim3 grdD((G3 / 16 + 15) / 16, TLEN / 16);
        gemm_nt_wmma<<<grdD, blk, 0, stream>>>(X + (size_t)NHIST * INP, INP, WdecT, INP,
                                               SCR + (size_t)NHIST * G3, G3, bi_dec,
                                               TLEN, G3, INP);
    }

    // 4. gate nonlinearity -> H f16 (decoder rows also packed into OUT left half)
    gru_gate_kernel<<<(NHIST * HID + T - 1) / T, T, 0, stream>>>(SCR, bh_enc, H, nullptr, NHIST);
    gru_gate_kernel<<<(TLEN * HID + T - 1) / T, T, 0, stream>>>(SCR + (size_t)NHIST * G3, bh_dec,
                                                                H + (size_t)NHIST * HID, OUT, TLEN);

    // 5. attention logits = hidden_state @ hidden_history^T  (NT form, K=512)
    {
        dim3 blk(128), grd((NHIST / 16 + 15) / 16, TLEN / 16);
        gemm_nt_wmma<<<grd, blk, 0, stream>>>(H + (size_t)NHIST * HID, HID, H, HID,
                                              SCR, NHIST, nullptr, TLEN, NHIST, HID);
    }

    // 6. attn = softmax rows -> f16
    softmax_f32_to_f16<<<TLEN, 256, 0, stream>>>(SCR, ATTN, NHIST);

    // 7. HHT[n][k] = hidden_history[k][n] for NT context GEMM
    transpose_f16<<<(NHIST * HID + T - 1) / T, T, 0, stream>>>(H, HHT, NHIST, HID);

    // 8. context = attn @ hidden_history  (K=6144, N=512), f32 into SCR
    {
        dim3 blk(128), grd((HID / 16 + 15) / 16, TLEN / 16);
        gemm_nt_wmma<<<grd, blk, 0, stream>>>(ATTN, NHIST, HHT, NHIST,
                                              SCR, HID, nullptr, TLEN, HID, NHIST);
    }

    // 9. OUT right half = context f16
    pack_out_right<<<(TLEN * HID + T - 1) / T, T, 0, stream>>>(SCR, OUT);

    // 10. final FC: score_logits = OUT @ W_fc + b_fc  (M=2048, N=50000, K=1024) -> d_out
    {
        dim3 blk(128), grd((LOC_SZ / 16 + 15) / 16, TLEN / 16);
        gemm_nt_wmma<<<grd, blk, 0, stream>>>(OUT, OUT2, WfcT, OUT2,
                                              score, LOC_SZ, b_fc, TLEN, LOC_SZ, OUT2);
    }

    // 11. row softmax in place over 50000 columns
    softmax_f32_inplace<<<TLEN, 256, 0, stream>>>(score, LOC_SZ);
}
